// GCN_10591389352059
// MI455X (gfx1250) — matrix operand order repaired and checked by reading, not compiled, not run on hardware
//
#include <hip/hip_runtime.h>
#include <hip/hip_bf16.h>

#define N_NODES   100000
#define N_EDGES   1600000
#define IN_FEATS  128
#define N_HIDDEN  128
#define N_CLASSES 40

typedef __attribute__((ext_vector_type(16))) __bf16 v16bf;
typedef __attribute__((ext_vector_type(8)))  float  v8f;

// ---------------- zero fill (float4 grid-stride) ----------------
__global__ void gcn_zero_kernel(float4* __restrict__ p, long n4) {
  long i = (long)blockIdx.x * blockDim.x + threadIdx.x;
  long stride = (long)gridDim.x * blockDim.x;
  float4 z = make_float4(0.f, 0.f, 0.f, 0.f);
  for (; i < n4; i += stride) p[i] = z;
}

// ---------------- bias broadcast init: out[n][c] = b[c] ----------------
__global__ void gcn_bias_init_kernel(float* __restrict__ out,
                                     const float* __restrict__ b,
                                     long n, int C) {
  long i = (long)blockIdx.x * blockDim.x + threadIdx.x;
  long stride = (long)gridDim.x * blockDim.x;
  for (; i < n; i += stride) out[i] = b[i % C];
}

// ---------------- edge-parallel scatter-add (128-wide rows) ----------------
// one wave32 per edge; each lane moves 4 of the 128 features.
// f32 atomics with discarded return -> GLOBAL_ATOMIC_ADD_F32 (no-return).
__global__ void gcn_scatter_kernel(const float* __restrict__ h,
                                   const int* __restrict__ src,
                                   const int* __restrict__ dst,
                                   float* __restrict__ out, int n_edges) {
  int edge = blockIdx.x * (blockDim.x >> 5) + (threadIdx.x >> 5);
  int lane = threadIdx.x & 31;
  if (edge >= n_edges) return;
  int s = src[edge];
  int d = dst[edge];
  float4 v = ((const float4*)(h + (size_t)s * 128))[lane];
  float* op = out + (size_t)d * 128 + lane * 4;
  __hip_atomic_fetch_add(op + 0, v.x, __ATOMIC_RELAXED, __HIP_MEMORY_SCOPE_AGENT);
  __hip_atomic_fetch_add(op + 1, v.y, __ATOMIC_RELAXED, __HIP_MEMORY_SCOPE_AGENT);
  __hip_atomic_fetch_add(op + 2, v.z, __ATOMIC_RELAXED, __HIP_MEMORY_SCOPE_AGENT);
  __hip_atomic_fetch_add(op + 3, v.w, __ATOMIC_RELAXED, __HIP_MEMORY_SCOPE_AGENT);
}

// ---------------- edge-parallel scatter-add (40-wide rows) ----------------
// flat thread-per-(edge, float2): 20 float2 per edge, coalesced within row.
__global__ void gcn_scatter40_kernel(const float* __restrict__ g,
                                     const int* __restrict__ src,
                                     const int* __restrict__ dst,
                                     float* __restrict__ out, int n_edges) {
  long t = (long)blockIdx.x * blockDim.x + threadIdx.x;
  long total = (long)n_edges * 20;
  if (t >= total) return;
  int edge = (int)(t / 20);
  int pos = (int)(t - (long)edge * 20) * 2;
  int s = src[edge];
  int d = dst[edge];
  float2 v = *(const float2*)(g + (size_t)s * 40 + pos);
  float* op = out + (size_t)d * 40 + pos;
  __hip_atomic_fetch_add(op + 0, v.x, __ATOMIC_RELAXED, __HIP_MEMORY_SCOPE_AGENT);
  __hip_atomic_fetch_add(op + 1, v.y, __ATOMIC_RELAXED, __HIP_MEMORY_SCOPE_AGENT);
}

// ---------------- weight convert + transpose + pad to bf16 ----------------
// W is [K][N] f32 row-major; Wt is [Npad][K] bf16 (zero-padded columns).
__global__ void gcn_cvtw_kernel(const float* __restrict__ W,
                                __bf16* __restrict__ Wt,
                                int K, int N, int Npad) {
  int idx = blockIdx.x * blockDim.x + threadIdx.x;
  if (idx >= Npad * K) return;
  int n = idx / K;
  int k = idx - n * K;
  float v = (n < N) ? W[(size_t)k * N + n] : 0.f;
  Wt[idx] = (__bf16)v;
}

// ---------------- bf16 WMMA GEMM: out = act(A @ W [+ b]) ----------------
// A: [M][128] f32 activations. Bt: [NT*16][128] bf16 transposed weights.
// Each wave computes a 16(M) x NT*16(N) row-strip. Per K-step: A converted
// once, ALL NT B tiles preloaded into independent registers (clause-able),
// then NT back-to-back WMMAs drain the waits progressively.
template <int NT>
__global__ __launch_bounds__(128)
void gcn_gemm_kernel(const float* __restrict__ A,
                     const __bf16* __restrict__ Bt,
                     const float* __restrict__ bias,
                     float* __restrict__ out,
                     int Mtiles, int Nreal, int do_relu, int add_bias) {
  const int K = 128;
  int wave = threadIdx.x >> 5;
  int lane = threadIdx.x & 31;
  int mtile = blockIdx.x * (blockDim.x >> 5) + wave;
  if (mtile >= Mtiles) return;

  int g  = lane >> 4;   // lane group 0/1
  int mn = lane & 15;   // row for A-operand, col for B/C/D operands

  const float* arow = A + (size_t)(mtile * 16 + mn) * K;

  v8f acc[NT];
#pragma unroll
  for (int nt = 0; nt < NT; ++nt) {
    int colN = nt * 16 + mn;
    float bv = (add_bias && colN < Nreal) ? bias[colN] : 0.f;
#pragma unroll
    for (int i = 0; i < 8; ++i) acc[nt][i] = bv;
  }

#pragma unroll
  for (int kb = 0; kb < K; kb += 32) {
    // A 16x32 bf16 operand: lane holds K = kb + g*16 + [0..15], contiguous
    const float4* a0 = (const float4*)(arow + kb + g * 16);
    const float4* a1 = (const float4*)(arow + kb + g * 16 + 8);
    float4 p0 = a0[0], p1 = a0[1];
    float4 q0 = a1[0], q1 = a1[1];
    v16bf a;
    a[0]  = (__bf16)p0.x; a[1]  = (__bf16)p0.y; a[2]  = (__bf16)p0.z; a[3]  = (__bf16)p0.w;
    a[4]  = (__bf16)p1.x; a[5]  = (__bf16)p1.y; a[6]  = (__bf16)p1.z; a[7]  = (__bf16)p1.w;
    a[8]  = (__bf16)q0.x; a[9]  = (__bf16)q0.y; a[10] = (__bf16)q0.z; a[11] = (__bf16)q0.w;
    a[12] = (__bf16)q1.x; a[13] = (__bf16)q1.y; a[14] = (__bf16)q1.z; a[15] = (__bf16)q1.w;

    // B 32x16 bf16 operands: lane holds K = kb + g*16 + [0..15], contiguous.
    // Preload all NT tiles so loads can issue together and overlap the WMMAs.
    v16bf b[NT];
#pragma unroll
    for (int nt = 0; nt < NT; ++nt)
      b[nt] = *(const v16bf*)(Bt + (size_t)(nt * 16 + mn) * K + g * 16 + kb);

#pragma unroll
    for (int nt = 0; nt < NT; ++nt)
      acc[nt] = __builtin_amdgcn_wmma_f32_16x16x32_bf16(
          /*neg_a=*/false, a, /*neg_b=*/false, b[nt],
          /*c_mod=*/(short)0, acc[nt], /*reuse_a=*/false, /*reuse_b=*/false);
  }

#pragma unroll
  for (int nt = 0; nt < NT; ++nt) {
    int colN = nt * 16 + mn;
    if (colN < Nreal) {
#pragma unroll
      for (int v = 0; v < 8; ++v) {
        int mr = mtile * 16 + v + 8 * g;
        float r = acc[nt][v];
        if (do_relu) r = fmaxf(r, 0.f);
        out[(size_t)mr * Nreal + colN] = r;
      }
    }
  }
}

// ---------------- host-side orchestration ----------------
extern "C" void kernel_launch(void* const* d_in, const int* in_sizes, int n_in,
                              void* d_out, int out_size, void* d_ws, size_t ws_size,
                              hipStream_t stream) {
  const float* features = (const float*)d_in[0];
  const int*   src      = (const int*)d_in[1];
  const int*   dst      = (const int*)d_in[2];
  const float* W1       = (const float*)d_in[3];
  const float* b1       = (const float*)d_in[4];
  const float* W2       = (const float*)d_in[5];
  const float* b2       = (const float*)d_in[6];
  const float* W3       = (const float*)d_in[7];
  const float* b3       = (const float*)d_in[8];
  float* out = (float*)d_out;

  const int NPAD3 = 48; // N_CLASSES padded to a multiple of 16

  // workspace layout (256B-aligned slices)
  char* ws = (char*)d_ws;
  size_t offA  = 0;                                   // agg buffer  [100000][128] f32
  size_t offB  = offA + (size_t)N_NODES * 128 * 4;    // act buffer  [100000][128] f32
  size_t offW1 = offB + (size_t)N_NODES * 128 * 4;    // Wt1 bf16 [128][128]
  size_t offW2 = offW1 + ((size_t)128 * 128 * 2 + 255) / 256 * 256;
  size_t offW3 = offW2 + ((size_t)128 * 128 * 2 + 255) / 256 * 256;
  float*  bufA = (float*)(ws + offA);
  float*  bufB = (float*)(ws + offB);
  __bf16* Wt1  = (__bf16*)(ws + offW1);
  __bf16* Wt2  = (__bf16*)(ws + offW2);
  __bf16* Wt3  = (__bf16*)(ws + offW3);

  // weight prep (bf16 transpose, pad)
  {
    int n1 = 128 * 128;
    gcn_cvtw_kernel<<<(n1 + 255) / 256, 256, 0, stream>>>(W1, Wt1, 128, N_HIDDEN, 128);
    gcn_cvtw_kernel<<<(n1 + 255) / 256, 256, 0, stream>>>(W2, Wt2, 128, N_HIDDEN, 128);
    int n3 = NPAD3 * 128;
    gcn_cvtw_kernel<<<(n3 + 255) / 256, 256, 0, stream>>>(W3, Wt3, 128, N_CLASSES, NPAD3);
  }

  long n4 = (long)N_NODES * 128 / 4;
  int edges_per_block = 256 / 32;
  int sgrid = (N_EDGES + edges_per_block - 1) / edges_per_block;

  const int MTILES = N_NODES / 16;          // 6250
  int ggrid = (MTILES + 3) / 4;             // 4 waves (M-tiles) per block

  // layer 1: agg(features) -> relu(@W1+b1)
  gcn_zero_kernel<<<1024, 256, 0, stream>>>((float4*)bufA, n4);
  gcn_scatter_kernel<<<sgrid, 256, 0, stream>>>(features, src, dst, bufA, N_EDGES);
  gcn_gemm_kernel<8><<<ggrid, 128, 0, stream>>>(bufA, Wt1, b1, bufB, MTILES, N_HIDDEN, 1, 1);

  // layer 2: agg(h1) -> relu(@W2+b2)
  gcn_zero_kernel<<<1024, 256, 0, stream>>>((float4*)bufA, n4);
  gcn_scatter_kernel<<<sgrid, 256, 0, stream>>>(bufB, src, dst, bufA, N_EDGES);
  gcn_gemm_kernel<8><<<ggrid, 128, 0, stream>>>(bufA, Wt2, b2, bufB, MTILES, N_HIDDEN, 1, 1);

  // layer 3 (reordered: GEMM first, then aggregate the narrow 40-dim rows):
  //   g = h2 @ W3      (no bias, no relu)  -> bufA [100000][40]
  //   out = b3 + agg(g)                    (bias via init, scatter-add f32)
  gcn_gemm_kernel<3><<<ggrid, 128, 0, stream>>>(bufB, Wt3, b3, bufA, MTILES, N_CLASSES, 0, 0);
  long nout = (long)N_NODES * N_CLASSES;
  gcn_bias_init_kernel<<<1024, 256, 0, stream>>>(out, b3, nout, N_CLASSES);
  long s40 = (long)N_EDGES * 20;
  gcn_scatter40_kernel<<<(int)((s40 + 255) / 256), 256, 0, stream>>>(bufA, src, dst, out, N_EDGES);
}
